// RNNModel_10368051053183
// MI455X (gfx1250) — matrix operand order repaired and checked by reading, not compiled, not run on hardware
//
#include <hip/hip_runtime.h>

typedef __attribute__((ext_vector_type(16))) __bf16        v16bf;
typedef __attribute__((ext_vector_type(8)))  __bf16        v8bf;
typedef __attribute__((ext_vector_type(8)))  float         v8f;
typedef __attribute__((ext_vector_type(8)))  unsigned int  v8u;

namespace {
constexpr int Bz   = 128;   // batch
constexpr int Tz   = 1024;  // seq len
constexpr int Mz   = 64;    // output dim
constexpr int Kz   = 256;   // hidden dim
constexpr int XD   = 65;    // input feature dim (M+1)
constexpr int XDP  = 96;    // padded to 3 K-chunks of 32
constexpr int ROWS = 16;    // batch rows per workgroup
constexpr int NTHR = 512;   // 16 waves of 32
constexpr int SH   = Kz  + 8;  // 264: padded LDS row stride (bank-conflict-free b128)
constexpr int SX   = XDP + 8;  // 104
constexpr int SV   = Kz  + 8;  // 264
}

union Frag { v16bf v; v8bf h[2]; };

// Round-to-nearest-even pack of two f32 into one dword of two bf16.
__device__ __forceinline__ unsigned int bf16pack(float a, float b) {
  union { float f; unsigned int u; } ua, ub;
  ua.f = a; ub.f = b;
  unsigned int x = (ua.u + 0x7fffu + ((ua.u >> 16) & 1u)) >> 16;
  unsigned int y = (ub.u + 0x7fffu + ((ub.u >> 16) & 1u)) >> 16;
  return x | (y << 16);
}

// Force the packed fragment dwords into VGPRs and make them opaque, so the
// compiler keeps them resident across the scan loop instead of
// re-materializing the f32->bf16 packing every iteration.
__device__ __forceinline__ void pin_frag(v8u& f) {
#pragma unroll
  for (int j = 0; j < 8; ++j) {
    unsigned int x = f[j];
    asm("" : "+v"(x));
    f[j] = x;
  }
}

__device__ __forceinline__ float fast_tanh(float x) {
#if __has_builtin(__builtin_amdgcn_tanhf)
  return __builtin_amdgcn_tanhf(x);
#else
  return tanhf(x);
#endif
}

__device__ __forceinline__ v16bf ld_frag(const __bf16* p0, const __bf16* p1) {
  Frag f;
  f.h[0] = *(const v8bf*)p0;
  f.h[1] = *(const v8bf*)p1;
  return f.v;
}

__device__ __forceinline__ v8f wmma_bf16(v16bf a, v16bf b, v8f c) {
  return __builtin_amdgcn_wmma_f32_16x16x32_bf16(false, a, false, b, (short)0, c,
                                                 false, false);
}
__device__ __forceinline__ v8f wmma_bf16u(v16bf a, v8u b, v8f c) {
  return __builtin_amdgcn_wmma_f32_16x16x32_bf16(false, a, false,
                                                 __builtin_bit_cast(v16bf, b),
                                                 (short)0, c, false, false);
}

__global__ __launch_bounds__(NTHR, 1)
void rnn_fused_scan(const float* __restrict__ xin,  // [B,T,65]
                    const float* __restrict__ Wih,  // [65,256]
                    const float* __restrict__ Whh,  // [256,256]
                    const float* __restrict__ Vw,   // [64,256]
                    float* __restrict__ out)        // [B,T,64]
{
  __shared__ __bf16 hbuf[2][ROWS * SH];   // ping-pong hidden state (bf16)
  __shared__ __bf16 xbuf[2][ROWS * SX];   // staged x_t (bf16, zero-padded to 96)
  __shared__ __bf16 vbuf[Mz * SV];        // V_w in bf16, row-major

  const int tid    = threadIdx.x;
  const int wave   = tid >> 5;        // 0..15
  const int lane   = tid & 31;
  const int l16    = lane & 15;
  const int lhalf  = lane >> 4;       // 0 or 1
  const int bstart = blockIdx.x * ROWS;

  // ---- init: V_w -> LDS bf16 ----
  for (int i = tid; i < Mz * Kz; i += NTHR) {
    int r = i >> 8, c = i & (Kz - 1);
    vbuf[r * SV + c] = (__bf16)Vw[i];
  }
  // ---- init: h_{-1} = 0 in buffer 0 ----
  for (int i = tid; i < ROWS * SH; i += NTHR) hbuf[0][i] = (__bf16)0.0f;

  // ---- register-resident W_hh B-fragments as PACKED dword vectors ----
  // B layout: lane n = l%16; lanes 0-15 hold K 0..15 of the chunk, lanes 16-31 K 16..31.
  // Packed dword j of lane l = { bf16(W[kb+2j][n]), bf16(W[kb+2j+1][n]) }.
  const int ncol = wave * 16 + l16;
  v8u whh[8];
#pragma unroll
  for (int c = 0; c < 8; ++c) {
    v8u t;
    int kb = c * 32 + lhalf * 16;
#pragma unroll
    for (int j = 0; j < 8; ++j)
      t[j] = bf16pack(Whh[(kb + 2 * j) * Kz + ncol],
                      Whh[(kb + 2 * j + 1) * Kz + ncol]);
    pin_frag(t);
    whh[c] = t;
  }
  // ---- register-resident W_ih B-fragments (K padded 65 -> 96), packed ----
  v8u wih[3];
#pragma unroll
  for (int c = 0; c < 3; ++c) {
    v8u t;
    int kb = c * 32 + lhalf * 16;
#pragma unroll
    for (int j = 0; j < 8; ++j) {
      int k0 = kb + 2 * j, k1 = kb + 2 * j + 1;
      float f0 = (k0 < XD) ? Wih[k0 * Kz + ncol] : 0.0f;
      float f1 = (k1 < XD) ? Wih[k1 * Kz + ncol] : 0.0f;
      t[j] = bf16pack(f0, f1);
    }
    pin_frag(t);
    wih[c] = t;
  }

  // ---- x staging ownership: 3 (row,col) slots per thread, prefetch t=0 ----
  int xrow[3], xcol[3];
  float xg[3];
#pragma unroll
  for (int s = 0; s < 3; ++s) {
    int p = tid + s * NTHR;           // 0 .. 16*96-1
    xrow[s] = p / XDP;
    xcol[s] = p % XDP;
    xg[s] = 0.0f;
    if (xcol[s] < XD)
      xg[s] = xin[((bstart + xrow[s]) * Tz + 0) * XD + xcol[s]];
  }

  // A-fragment base offsets (per lane).
  // A layout: lane row m = l%16; elems 0..15 = K kb..kb+15,
  // where kb = 32c + (lane>=16 ? 16 : 0).
  const int aoff = l16 * SH + lhalf * 16;
  const int xoff = l16 * SX + lhalf * 16;
  const int voff = ((wave & 3) * 16 + l16) * SV + lhalf * 16;  // y B-frags (waves 0-3)

  __syncthreads();

  for (int t = 0; t < Tz; ++t) {
    const int q = t & 1;
    // stage x_t (prefetched registers) -> LDS
#pragma unroll
    for (int s = 0; s < 3; ++s)
      xbuf[q][xrow[s] * SX + xcol[s]] = (__bf16)xg[s];
    __syncthreads();   // xbuf[q] ready; hbuf[q] (= h_{t-1}) ready from previous iter

    // prefetch x_{t+1} from global (hidden under the WMMA work below)
    if (t + 1 < Tz) {
#pragma unroll
      for (int s = 0; s < 3; ++s)
        if (xcol[s] < XD)
          xg[s] = xin[((bstart + xrow[s]) * Tz + (t + 1)) * XD + xcol[s]];
    }

    const bool doY = (wave < 4) && (t > 0);   // y_{t-1} piggybacks on h_{t-1} frags
    v8f acc0 = {}, acc1 = {};
    v8f y0 = {}, y1 = {};

#pragma unroll
    for (int c = 0; c < 8; ++c) {
      v16bf a = ld_frag(&hbuf[q][aoff + 32 * c], &hbuf[q][aoff + 32 * c + 8]);
      if (c & 1) acc1 = wmma_bf16u(a, whh[c], acc1);
      else       acc0 = wmma_bf16u(a, whh[c], acc0);
      if (doY) {
        v16bf b = ld_frag(&vbuf[voff + 32 * c], &vbuf[voff + 32 * c + 8]);
        if (c & 1) y1 = wmma_bf16(a, b, y1);
        else       y0 = wmma_bf16(a, b, y0);
      }
    }
#pragma unroll
    for (int c = 0; c < 3; ++c) {
      v16bf a = ld_frag(&xbuf[q][xoff + 32 * c], &xbuf[q][xoff + 32 * c + 8]);
      if (c & 1) acc1 = wmma_bf16u(a, wih[c], acc1);
      else       acc0 = wmma_bf16u(a, wih[c], acc0);
    }

    if (doY) {
      // D layout: elem j, lane l -> row (batch) = j + (l>=16)*8, col (m) = 16w + l%16
#pragma unroll
      for (int j = 0; j < 8; ++j) {
        int brow = bstart + j + lhalf * 8;
        out[((size_t)brow * Tz + (t - 1)) * Mz + wave * 16 + l16] = y0[j] + y1[j];
      }
    }

    // h_t = tanh(acc); store bf16 into the other buffer
#pragma unroll
    for (int j = 0; j < 8; ++j) {
      float hv = fast_tanh(acc0[j] + acc1[j]);
      hbuf[q ^ 1][(j + lhalf * 8) * SH + ncol] = (__bf16)hv;
    }
  }

  // ---- epilogue: y_{T-1} (h_{T-1} lives in hbuf[Tz & 1]) ----
  __syncthreads();
  if (wave < 4) {
    const int q = Tz & 1;
    v8f y0 = {}, y1 = {};
#pragma unroll
    for (int c = 0; c < 8; ++c) {
      v16bf a = ld_frag(&hbuf[q][aoff + 32 * c], &hbuf[q][aoff + 32 * c + 8]);
      v16bf b = ld_frag(&vbuf[voff + 32 * c], &vbuf[voff + 32 * c + 8]);
      if (c & 1) y1 = wmma_bf16(a, b, y1);
      else       y0 = wmma_bf16(a, b, y0);
    }
#pragma unroll
    for (int j = 0; j < 8; ++j) {
      int brow = bstart + j + lhalf * 8;
      out[((size_t)brow * Tz + (Tz - 1)) * Mz + wave * 16 + l16] = y0[j] + y1[j];
    }
  }
}

extern "C" void kernel_launch(void* const* d_in, const int* in_sizes, int n_in,
                              void* d_out, int out_size, void* d_ws, size_t ws_size,
                              hipStream_t stream) {
  (void)in_sizes; (void)n_in; (void)out_size; (void)d_ws; (void)ws_size;
  const float* xin = (const float*)d_in[0];   // [128,1024,65]
  const float* Wih = (const float*)d_in[1];   // [65,256]
  const float* Whh = (const float*)d_in[2];   // [256,256]
  const float* Vw  = (const float*)d_in[3];   // [64,256]
  float* out = (float*)d_out;                 // [128,1024,64]

  rnn_fused_scan<<<dim3(Bz / ROWS), dim3(NTHR), 0, stream>>>(xin, Wih, Whh, Vw, out);
}
